// MultiHeadAttention_36404142801315
// MI455X (gfx1250) — compile-verified, matrix operand order checked by reading, not run on hardware
//
#include <hip/hip_runtime.h>
#include <hip/hip_bf16.h>

// ---------------------------------------------------------------------------
// MHA forward on gfx1250: bf16 WMMA everywhere, async-to-LDS DMA (CDNA5)
// with LDS double-buffering in the GEMMs, flash-style online-softmax attention.
// ---------------------------------------------------------------------------

typedef __bf16 bf16;
typedef __attribute__((ext_vector_type(16))) __bf16 v16bf;
typedef __attribute__((ext_vector_type(8)))  float  v8f;

#define DIMC   1024
#define NHEAD  16
#define HDIM   64
#define SEQ    2048
#define BATCH  2
#define ROWS   (BATCH * SEQ)   // 4096
#define QKVN   (3 * DIMC)      // 3072
#define ATTN_SCALE 0.125f      // 64^-0.5

union Frag16 { v16bf v; uint4 q[2]; bf16 e[16]; };
union Chunk8 { uint4 q; bf16 e[8]; };

// --- CDNA5 async memory->LDS DMA (ASYNCcnt-tracked, no VGPR round trip) ----
typedef __attribute__((address_space(3))) void lds_void;

__device__ __forceinline__ unsigned to_lds_off(const void* p) {
    // addrspace(3) pointer values are byte offsets from the wave's LDS base,
    // which is exactly what GLOBAL_LOAD_ASYNC_TO_LDS expects in VDST.
    return (unsigned)(unsigned long long)(lds_void*)p;
}

__device__ __forceinline__ void async_load_b128(void* lds_dst, const void* gsrc) {
    asm volatile("global_load_async_to_lds_b128 %0, %1, off"
                 :: "v"(to_lds_off(lds_dst)), "v"(gsrc)
                 : "memory");
}

#define S_WAIT_ASYNCCNT(N) asm volatile("s_wait_asynccnt %0" :: "n"(N) : "memory")

// ---------------------------------------------------------------------------
// fp32 -> bf16 conversion (vectorized 4-wide)
// ---------------------------------------------------------------------------
__global__ __launch_bounds__(256)
void cvt_f32_bf16(const float* __restrict__ in, bf16* __restrict__ out, int n) {
    int i = (blockIdx.x * blockDim.x + threadIdx.x) * 4;
    if (i >= n) return;
    float4 f = *(const float4*)(in + i);
    union { bf16 e[4]; uint2 u; } p;
    p.e[0] = (bf16)f.x; p.e[1] = (bf16)f.y; p.e[2] = (bf16)f.z; p.e[3] = (bf16)f.w;
    *(uint2*)(out + i) = p.u;
}

// ---------------------------------------------------------------------------
// C[M,N] = A[M,K] (bf16) x W[N,K]^T (bf16) + bias[N]
// Block: 128 threads (4 waves), tile 128(M) x 64(N), K-step 32.
// Each wave computes a 32x64 strip: 2 A-frags x 4 B-frags = 8 wmma / K-step.
// Global->LDS staging uses async DMA + double-buffered LDS: tile i+1 streams
// in while tile i feeds the WMMAs. 6 async b128 per thread per tile.
// ---------------------------------------------------------------------------
#define GLDA 40  // LDS row stride in bf16 (80B: 16B-aligned, bank-skewed)

__global__ __launch_bounds__(128)
void gemm_bf16_wmma(const bf16* __restrict__ A,   // [M,K]
                    const bf16* __restrict__ W,   // [N,K]
                    const float* __restrict__ bias,
                    float* __restrict__ Cf,       // fp32 out (may be null)
                    bf16* __restrict__ Cbf,       // bf16 out (may be null)
                    int M, int N, int K) {
    __shared__ alignas(16) bf16 sA[2][128 * GLDA];
    __shared__ alignas(16) bf16 sW[2][64 * GLDA];

    const int t    = threadIdx.x;
    const int wave = t >> 5;
    const int lane = t & 31;
    const int half = lane >> 4;
    const int lx   = lane & 15;
    const int m0   = blockIdx.y * 128;
    const int n0   = blockIdx.x * 64;

    v8f acc[2][4] = {};

    // Issue async loads for one 128x32 A tile + 64x32 W tile into buffer `buf`
    auto issue_tile = [&](int kt, int buf) {
        // A: 128x32 bf16 = 512 chunks of 8; 4 per thread
        for (int i = 0; i < 4; ++i) {
            int idx = t + i * 128;
            int row = idx >> 2;
            int col = (idx & 3) * 8;
            async_load_b128(&sA[buf][row * GLDA + col],
                            &A[(size_t)(m0 + row) * K + kt + col]);
        }
        // W: 64x32 bf16 = 256 chunks of 8; 2 per thread
        for (int i = 0; i < 2; ++i) {
            int idx = t + i * 128;
            int row = idx >> 2;
            int col = (idx & 3) * 8;
            async_load_b128(&sW[buf][row * GLDA + col],
                            &W[(size_t)(n0 + row) * K + kt + col]);
        }
    };

    issue_tile(0, 0);  // prologue: tile 0 -> buffer 0

    int it = 0;
    for (int kt = 0; kt < K; kt += 32, ++it) {
        const int buf = it & 1;
        __syncthreads();                 // everyone done reading buf^1
        if (kt + 32 < K) {
            issue_tile(kt + 32, buf ^ 1);
            S_WAIT_ASYNCCNT(6);          // this tile's 6 ops retired (in-order)
        } else {
            S_WAIT_ASYNCCNT(0);
        }
        __syncthreads();                 // tile data visible to all waves

        const bf16* sAb = &sA[buf][0];
        const bf16* sWb = &sW[buf][0];

        Frag16 fb[4];
        for (int nt = 0; nt < 4; ++nt) {
            const bf16* wr = &sWb[(nt * 16 + lx) * GLDA];
            fb[nt].q[0] = *(const uint4*)(wr + 8 * half);
            fb[nt].q[1] = *(const uint4*)(wr + 16 + 8 * half);
        }
        for (int ms = 0; ms < 2; ++ms) {
            Frag16 fa;
            const bf16* ar = &sAb[(wave * 32 + ms * 16 + lx) * GLDA];
            fa.q[0] = *(const uint4*)(ar + 8 * half);
            fa.q[1] = *(const uint4*)(ar + 16 + 8 * half);
            for (int nt = 0; nt < 4; ++nt) {
                acc[ms][nt] = __builtin_amdgcn_wmma_f32_16x16x32_bf16(
                    false, fa.v, false, fb[nt].v, (short)0, acc[ms][nt],
                    false, false);
            }
        }
    }

    // Epilogue: bias + store
    for (int ms = 0; ms < 2; ++ms) {
        for (int nt = 0; nt < 4; ++nt) {
            int col = n0 + nt * 16 + lx;
            float bv = bias[col];
            for (int r = 0; r < 8; ++r) {
                int row = m0 + wave * 32 + ms * 16 + r + 8 * half;
                float v = acc[ms][nt][r] + bv;
                if (Cf)  Cf[(size_t)row * N + col] = v;
                if (Cbf) Cbf[(size_t)row * N + col] = (bf16)v;
            }
        }
    }
}

// ---------------------------------------------------------------------------
// Flash attention: grid (SEQ/64, BATCH*NHEAD), 128 threads (4 waves).
// Each wave owns 16 query rows; all waves sweep the same 32-key tiles.
// V chunks for tile kt+32 are prefetched into registers right after the
// current tile's transposed LDS stores, overlapping VMEM with wmma/softmax.
// ---------------------------------------------------------------------------
#define VLD 40  // transposed V tile row stride in bf16

__global__ __launch_bounds__(128)
void attn_wmma(const bf16* __restrict__ qkv, bf16* __restrict__ attn_out) {
    __shared__ alignas(16) bf16 sVt[HDIM * VLD];     // V^T tile: [d][key]
    __shared__ alignas(16) bf16 sP[4 * 16 * 32];     // per-wave P tile

    const int t    = threadIdx.x;
    const int wave = t >> 5;
    const int lane = t & 31;
    const int half = lane >> 4;
    const int lx   = lane & 15;
    const int bh   = blockIdx.y;
    const int b    = bh >> 4;
    const int h    = bh & 15;
    const int q0   = blockIdx.x * 64 + wave * 16;

    const bf16* Qb = qkv + (size_t)(b * SEQ) * QKVN + h * HDIM;
    const bf16* Kb = Qb + DIMC;
    const bf16* Vb = Qb + 2 * DIMC;

    // Persistent Q fragments: two 16x32 A-frags over d (0-31, 32-63)
    Frag16 aq[2];
    {
        const bf16* qr = Qb + (size_t)(q0 + lx) * QKVN;
        for (int c = 0; c < 2; ++c) {
            aq[c].q[0] = *(const uint4*)(qr + 32 * c + 8 * half);
            aq[c].q[1] = *(const uint4*)(qr + 32 * c + 16 + 8 * half);
        }
    }

    v8f o[4] = {};
    float m[8], l[8];
    for (int r = 0; r < 8; ++r) { m[r] = -1e30f; l[r] = 0.0f; }

    bf16* Pw = &sP[wave * 16 * 32];

    // Register-pipelined V chunks: 2 x 8 bf16 per thread per 32-key tile
    Chunk8 vch[2];
    auto fetch_v = [&](int kt) {
        for (int i = 0; i < 2; ++i) {
            int idx   = t + i * 128;
            int kk    = idx >> 3;
            int dbase = (idx & 7) * 8;
            vch[i].q = *(const uint4*)(Vb + (size_t)(kt + kk) * QKVN + dbase);
        }
    };
    fetch_v(0);

    for (int kt = 0; kt < SEQ; kt += 32) {
        __syncthreads();
        // Transposed store of current V tile into sVt[d][key]
        for (int i = 0; i < 2; ++i) {
            int idx   = t + i * 128;
            int kk    = idx >> 3;
            int dbase = (idx & 7) * 8;
            for (int j = 0; j < 8; ++j)
                sVt[(dbase + j) * VLD + kk] = vch[i].e[j];
        }
        __syncthreads();
        if (kt + 32 < SEQ) fetch_v(kt + 32);   // overlap with compute below

        // Scores: S[16q x 32keys] via 4 wmma (2 key subtiles x 2 d-chunks)
        v8f s[2] = {};
        for (int j = 0; j < 2; ++j) {
            const bf16* kr = Kb + (size_t)(kt + j * 16 + lx) * QKVN;
            for (int c = 0; c < 2; ++c) {
                Frag16 fbk;
                fbk.q[0] = *(const uint4*)(kr + 32 * c + 8 * half);
                fbk.q[1] = *(const uint4*)(kr + 32 * c + 16 + 8 * half);
                s[j] = __builtin_amdgcn_wmma_f32_16x16x32_bf16(
                    false, aq[c].v, false, fbk.v, (short)0, s[j], false, false);
            }
        }

        // Online softmax (rows live across 16-lane groups -> xor masks 1..8)
        for (int r = 0; r < 8; ++r) {
            float v0 = s[0][r] * ATTN_SCALE;
            float v1 = s[1][r] * ATTN_SCALE;
            float mx = fmaxf(v0, v1);
            for (int off = 8; off >= 1; off >>= 1)
                mx = fmaxf(mx, __shfl_xor(mx, off, 32));
            float mn    = fmaxf(m[r], mx);
            float alpha = __expf(m[r] - mn);
            float p0    = __expf(v0 - mn);
            float p1    = __expf(v1 - mn);
            float ps    = p0 + p1;
            for (int off = 8; off >= 1; off >>= 1)
                ps += __shfl_xor(ps, off, 32);
            l[r] = l[r] * alpha + ps;
            m[r] = mn;
            for (int dt = 0; dt < 4; ++dt) o[dt][r] *= alpha;
            int prow = r + 8 * half;
            Pw[prow * 32 + lx]      = (bf16)p0;
            Pw[prow * 32 + 16 + lx] = (bf16)p1;
        }

        // Re-read P in A-fragment layout (keys are the wmma K dim)
        Frag16 ap;
        const bf16* pr = &Pw[lx * 32];
        ap.q[0] = *(const uint4*)(pr + 8 * half);
        ap.q[1] = *(const uint4*)(pr + 16 + 8 * half);

        // O += P * V : 4 wmma over the four 16-wide d tiles
        for (int dt = 0; dt < 4; ++dt) {
            Frag16 fv;
            const bf16* vr = &sVt[(dt * 16 + lx) * VLD];
            fv.q[0] = *(const uint4*)(vr + 8 * half);
            fv.q[1] = *(const uint4*)(vr + 16 + 8 * half);
            o[dt] = __builtin_amdgcn_wmma_f32_16x16x32_bf16(
                false, ap.v, false, fv.v, (short)0, o[dt], false, false);
        }
    }

    // Normalize and write bf16 output [ROWS, DIMC]
    for (int dt = 0; dt < 4; ++dt) {
        int col = h * HDIM + dt * 16 + lx;
        for (int r = 0; r < 8; ++r) {
            int row   = b * SEQ + q0 + r + 8 * half;
            float inv = 1.0f / l[r];
            attn_out[(size_t)row * DIMC + col] = (bf16)(o[dt][r] * inv);
        }
    }
}

// ---------------------------------------------------------------------------
// Launch
// ---------------------------------------------------------------------------
extern "C" void kernel_launch(void* const* d_in, const int* in_sizes, int n_in,
                              void* d_out, int out_size, void* d_ws, size_t ws_size,
                              hipStream_t stream) {
    const float* x     = (const float*)d_in[0];
    const float* Wqkv  = (const float*)d_in[1];
    const float* bqkv  = (const float*)d_in[2];
    const float* Wproj = (const float*)d_in[3];
    const float* bproj = (const float*)d_in[4];
    float* out = (float*)d_out;

    // Workspace layout (48 MiB total)
    char* ws = (char*)d_ws;
    bf16* x_bf     = (bf16*)(ws);                       //  8 MiB: [4096,1024]
    bf16* wqkv_bf  = (bf16*)(ws + ((size_t)8  << 20));  //  6 MiB: [3072,1024]
    bf16* wproj_bf = (bf16*)(ws + ((size_t)14 << 20));  //  2 MiB: [1024,1024]
    bf16* qkv_bf   = (bf16*)(ws + ((size_t)16 << 20));  // 24 MiB: [4096,3072]
    bf16* attn_bf  = (bf16*)(ws + ((size_t)40 << 20));  //  8 MiB: [4096,1024]

    // 1) fp32 -> bf16 conversions
    {
        int n = ROWS * DIMC;
        cvt_f32_bf16<<<(n / 4 + 255) / 256, 256, 0, stream>>>(x, x_bf, n);
        n = QKVN * DIMC;
        cvt_f32_bf16<<<(n / 4 + 255) / 256, 256, 0, stream>>>(Wqkv, wqkv_bf, n);
        n = DIMC * DIMC;
        cvt_f32_bf16<<<(n / 4 + 255) / 256, 256, 0, stream>>>(Wproj, wproj_bf, n);
    }

    // 2) QKV projection: [4096,3072] = x_bf [4096,1024] x Wqkv^T + b
    {
        dim3 grid(QKVN / 64, ROWS / 128);
        gemm_bf16_wmma<<<grid, 128, 0, stream>>>(
            x_bf, wqkv_bf, bqkv, nullptr, qkv_bf, ROWS, QKVN, DIMC);
    }

    // 3) Attention
    {
        dim3 grid(SEQ / 64, BATCH * NHEAD);
        attn_wmma<<<grid, 128, 0, stream>>>(qkv_bf, attn_bf);
    }

    // 4) Output projection: out = attn_bf x Wproj^T + b  (fp32 out)
    {
        dim3 grid(DIMC / 64, ROWS / 128);
        gemm_bf16_wmma<<<grid, 128, 0, stream>>>(
            attn_bf, wproj_bf, bproj, out, nullptr, ROWS, DIMC, DIMC);
    }
}
